// Network_61924838473886
// MI455X (gfx1250) — matrix operand order repaired and checked by reading, not compiled, not run on hardware
//
#include <hip/hip_runtime.h>

// ---------------------------------------------------------------------------
// SLAYER SNN forward for MI455X (gfx1250, wave32).
// Conv / FC -> implicit-GEMM on v_wmma_f32_16x16x32_bf16, fully static shapes.
// psp (FIR) -> register-resident row kernel (streams HBM once).
// spike -> sequential refractory recurrence, coefficients folded to literals.
// ---------------------------------------------------------------------------

typedef __attribute__((ext_vector_type(16))) __bf16 v16bf;
typedef __attribute__((ext_vector_type(8)))  float  v8f;

// PSP_KERN[k] = (k/10)*exp(1 - k/10), k = 0..31  (THETA=10, TAU_SR=10)
static constexpr float PSP_KERN[32] = {
  0.0f,          0.245960311f, 0.445108186f, 0.604125812f,
  0.728847520f,  0.824360635f, 0.895094819f, 0.944901165f,
  0.977122206f,  0.994653826f, 1.000000000f, 0.995321160f,
  0.982476904f,  0.963063687f, 0.938448064f, 0.909795990f,
  0.878098618f,  0.844195016f, 0.808792135f, 0.772482353f,
  0.735758882f,  0.699029276f, 0.662627266f, 0.626823124f,
  0.591832713f,  0.557825400f, 0.524930947f, 0.493245515f,
  0.462836888f,  0.433748997f, 0.406005850f, 0.379614928f
};

// REF_KERN[k] = -20*(k+1)*exp(-k), k = 0..15
static constexpr float REF_KERN[16] = {
  -20.0f,          -14.7151776f,   -8.12011699f,   -3.98296547f,
  -1.83156389f,    -0.808553640f,  -0.347025305f,  -0.145901114f,
  -0.060383273f,   -0.024681961f,  -0.009987985f,  -0.004008408f,
  -0.001597495f,   -0.000632892f,  -0.000249459f,  -0.0000978887f
};

// ------------------------------ psp: causal FIR along T --------------------
// One thread per neuron row: load T samples once, emit T outputs from regs.
template<int T>
__global__ void psp_row_kernel(const float* __restrict__ x,
                               float* __restrict__ y, int neurons) {
  int i = blockIdx.x * blockDim.x + threadIdx.x;
  if (i >= neurons) return;
  const float* xp = x + (size_t)i * T;
  float* yp = y + (size_t)i * T;
  float xv[T];
#pragma unroll
  for (int t = 0; t < T; ++t) xv[t] = xp[t];
#pragma unroll
  for (int t = 0; t < T; ++t) {
    float acc = 0.0f;
    const int kmax = (t + 1 < 32) ? (t + 1) : 32;
#pragma unroll
    for (int k = 0; k < kmax; ++k)
      acc = fmaf(xv[t - k], PSP_KERN[k], acc);
    yp[t] = acc;
  }
}

// ------------------------------ spike: sequential refractory ---------------
template<int T>
__global__ void spike_kernel(const float* __restrict__ u, float* __restrict__ s,
                             int n_neurons) {
  int i = blockIdx.x * blockDim.x + threadIdx.x;
  if (i >= n_neurons) return;
  float buf[16];
#pragma unroll
  for (int k = 0; k < 16; ++k) buf[k] = 0.0f;
  const float* up = u + (size_t)i * T;
  float* sp = s + (size_t)i * T;
  for (int t = 0; t < T; ++t) {
    float v = up[t] + buf[0];
    float sv = (v >= 10.0f) ? 1.0f : 0.0f;
#pragma unroll
    for (int k = 0; k < 15; ++k) buf[k] = buf[k + 1];
    buf[15] = 0.0f;
#pragma unroll
    for (int k = 0; k < 16; ++k) buf[k] = fmaf(sv, REF_KERN[k], buf[k]);
    sp[t] = sv;
  }
}

// ------------------------------ pool: 2x2 sum * 1.1*theta/4 ----------------
template<int N, int C, int H, int W, int T>
__global__ void pool_kernel(const float* __restrict__ x, float* __restrict__ y) {
  constexpr int Ho = H / 2, Wo = W / 2;
  constexpr long long total = (long long)N * C * Ho * Wo * T;
  long long gid = (long long)blockIdx.x * blockDim.x + threadIdx.x;
  if (gid >= total) return;
  int t = (int)(gid % T); long long r = gid / T;
  int wo = (int)(r % Wo); r /= Wo;
  int ho = (int)(r % Ho); r /= Ho;
  int c  = (int)(r % C);
  int n  = (int)(r / C);
  size_t base = (((size_t)(n * C + c) * H + 2 * ho) * W + 2 * wo) * T + t;
  constexpr size_t rowT = (size_t)W * T;
  float acc = x[base] + x[base + T] + x[base + rowT] + x[base + rowT + T];
  y[gid] = 2.75f * acc;   // 1.1 * 10 / 4
}

// ------------------------------ conv (and FC) as WMMA implicit GEMM --------
// y[n,co,ho,wo,t] = sum_{c,ki,kj} x[n,c,ho*S-P+ki, wo*S-P+kj, t] * w[co,c,ki,kj]
// A = W [Co x K], B = im2col [K x pixels], pixels linearized (n,ho,wo,t).
template<int N, int C, int H, int W, int T,
         int Co, int KH, int KW, int STRIDE, int PAD, int Ho, int Wo>
__global__ void conv_wmma_kernel(const float* __restrict__ x,
                                 const float* __restrict__ w,
                                 float* __restrict__ y) {
  constexpr int Ktot = C * KH * KW;
  constexpr int khkw = KH * KW;
  constexpr int totalPix = N * Ho * Wo * T;
  constexpr int pixTiles = (totalPix + 15) >> 4;
  constexpr int coTiles  = (Co + 15) >> 4;

  __shared__ float wlds[Co * Ktot];
  for (int i = threadIdx.x; i < Co * Ktot; i += blockDim.x) wlds[i] = w[i];
  __syncthreads();

  const int lane = threadIdx.x & 31;
  const int wid  = threadIdx.x >> 5;
  const int wavesPerBlock = blockDim.x >> 5;
  const int tile = blockIdx.x * wavesPerBlock + wid;
  if (tile >= pixTiles * coTiles) return;      // wave-uniform: EXEC stays full
  const int pixTile = tile % pixTiles;
  const int coTile  = tile / pixTiles;
  const int pixBase = pixTile << 4;

  // this lane's output pixel (B/D column index = lane & 15)
  const int q = pixBase + (lane & 15);
  const bool qv = (q < totalPix);
  int t = 0, wo = 0, ho = 0, n = 0;
  if (qv) {
    t = q % T; int r = q / T;
    wo = r % Wo; r /= Wo;
    ho = r % Ho; n  = r / Ho;
  }
  const int half = lane >> 4;
  const int co0  = (coTile << 4) + (lane & 15);   // A row this lane feeds
  const bool coAv = (co0 < Co);

  v8f acc = {0.f, 0.f, 0.f, 0.f, 0.f, 0.f, 0.f, 0.f};

#pragma unroll 4
  for (int kc = 0; kc < Ktot; kc += 32) {
    v16bf afrag, bfrag;
#pragma unroll
    for (int i = 0; i < 16; ++i) {
      const int v = i >> 1, sgn = i & 1;
      // A k order used here (contiguous k = 16*half + i): lanes 0-15 K=0..15, lanes 16-31 K=16..31
      const int kA = (half << 4) + v * 2 + sgn;
      int kk = kc + kA;
      float av = 0.0f;
      if (coAv && kk < Ktot) av = wlds[co0 * Ktot + kk];
      afrag[i] = (__bf16)av;
      // B k order used here (matches A, k = 16*half + i): lanes 0-15 K=0..15, lanes 16-31 K=16..31
      const int kB = (half << 4) + v * 2 + sgn;
      kk = kc + kB;
      float bv = 0.0f;
      if (qv && kk < Ktot) {
        const int c  = kk / khkw;
        const int rr = kk - c * khkw;
        const int ki = rr / KW;
        const int kj = rr - ki * KW;
        const int hh = ho * STRIDE - PAD + ki;
        const int ww = wo * STRIDE - PAD + kj;
        if (hh >= 0 && hh < H && ww >= 0 && ww < W)
          bv = x[(((size_t)(n * C + c) * H + hh) * W + ww) * T + t];
      }
      bfrag[i] = (__bf16)bv;
    }
    acc = __builtin_amdgcn_wmma_f32_16x16x32_bf16(false, afrag, false, bfrag,
                                                  (short)0, acc, false, false);
  }

  if (qv) {
#pragma unroll
    for (int r = 0; r < 8; ++r) {
      const int co = (coTile << 4) + r + (half << 3);   // D: M = r + 8*half
      if (co < Co)
        y[(((size_t)(n * Co + co) * Ho + ho) * Wo + wo) * T + t] = acc[r];
    }
  }
}

// ------------------------------ host-side launch helpers -------------------
template<int T>
static void launch_psp(const float* x, float* y, int neurons, hipStream_t s) {
  unsigned grid = (unsigned)((neurons + 255) / 256);
  psp_row_kernel<T><<<grid, 256, 0, s>>>(x, y, neurons);
}

template<int T>
static void launch_spike(const float* u, float* sout, int neurons, hipStream_t s) {
  unsigned grid = (unsigned)((neurons + 255) / 256);
  spike_kernel<T><<<grid, 256, 0, s>>>(u, sout, neurons);
}

template<int N, int C, int H, int W, int T>
static void launch_pool(const float* x, float* y, hipStream_t s) {
  constexpr long long total = (long long)N * C * (H / 2) * (W / 2) * T;
  unsigned grid = (unsigned)((total + 255) / 256);
  pool_kernel<N, C, H, W, T><<<grid, 256, 0, s>>>(x, y);
}

template<int N, int C, int H, int W, int T,
         int Co, int KH, int KW, int S, int P, int Ho, int Wo>
static void launch_conv(const float* x, const float* w, float* y, hipStream_t st) {
  constexpr int totalPix = N * Ho * Wo * T;
  constexpr int pixTiles = (totalPix + 15) / 16;
  constexpr int coTiles  = (Co + 15) / 16;
  constexpr int waves    = pixTiles * coTiles;
  constexpr unsigned blocks = (unsigned)((waves + 3) / 4);   // 4 waves / block
  conv_wmma_kernel<N, C, H, W, T, Co, KH, KW, S, P, Ho, Wo>
      <<<blocks, 128, 0, st>>>(x, w, y);
}

// ------------------------------ pipeline -----------------------------------
extern "C" void kernel_launch(void* const* d_in, const int* in_sizes, int n_in,
                              void* d_out, int out_size, void* d_ws, size_t ws_size,
                              hipStream_t stream) {
  (void)in_sizes; (void)n_in; (void)out_size; (void)ws_size;
  const float* in  = (const float*)d_in[0];   // [4,2,256,256,50]
  const float* w1  = (const float*)d_in[1];   // [8,2,5,5]
  const float* w2  = (const float*)d_in[2];   // [16,8,3,3]
  const float* w3  = (const float*)d_in[3];   // [32,16,3,3]
  const float* wfc = (const float*)d_in[4];   // [2,32,8,8]
  float* out = (float*)d_out;                 // [4,2,1,1,50] = 400

  const long long CAP = 26214400;             // max stage size (floats)
  float* A = (float*)d_ws;
  float* B = A + CAP;

  // L1: psp -> conv1 -> spike
  launch_psp<50>(in, A, 4 * 2 * 256 * 256, stream);                        // in -> A
  launch_conv<4, 2, 256, 256, 50, 8, 5, 5, 2, 1, 127, 127>(A, w1, B, stream);
  launch_spike<50>(B, A, 4 * 8 * 127 * 127, stream);                       // B -> A
  // L2: psp -> pool -> spike
  launch_psp<50>(A, B, 4 * 8 * 127 * 127, stream);                         // A -> B
  launch_pool<4, 8, 127, 127, 50>(B, A, stream);                           // B -> A
  launch_spike<50>(A, B, 4 * 8 * 63 * 63, stream);                         // A -> B
  // L3: psp -> conv2 -> spike
  launch_psp<50>(B, A, 4 * 8 * 63 * 63, stream);                           // B -> A
  launch_conv<4, 8, 63, 63, 50, 16, 3, 3, 2, 1, 32, 32>(A, w2, B, stream);
  launch_spike<50>(B, A, 4 * 16 * 32 * 32, stream);                        // B -> A
  // L4: psp -> pool -> spike
  launch_psp<50>(A, B, 4 * 16 * 32 * 32, stream);                          // A -> B
  launch_pool<4, 16, 32, 32, 50>(B, A, stream);                            // B -> A
  launch_spike<50>(A, B, 4 * 16 * 16 * 16, stream);                        // A -> B
  // L5: psp -> conv3 -> spike
  launch_psp<50>(B, A, 4 * 16 * 16 * 16, stream);                          // B -> A
  launch_conv<4, 16, 16, 16, 50, 32, 3, 3, 2, 1, 8, 8>(A, w3, B, stream);
  launch_spike<50>(B, A, 4 * 32 * 8 * 8, stream);                          // B -> A
  // FC: psp -> dense (as 8x8 conv, Ho=Wo=1, K=2048) -> spike
  launch_psp<50>(A, B, 4 * 32 * 8 * 8, stream);                            // A -> B
  launch_conv<4, 32, 8, 8, 50, 2, 8, 8, 1, 0, 1, 1>(B, wfc, A, stream);
  launch_spike<50>(A, out, 4 * 2, stream);                                 // A -> out
}
